// DSLPostProcessor_54906861912218
// MI455X (gfx1250) — compile-verified, matrix operand order checked by reading, not run on hardware
//
#include <hip/hip_runtime.h>
#include <hip/hip_bf16.h>
#include <stdint.h>
#include <math.h>

// ---------------- problem constants (from reference) ----------------
#define NIMG   8
#define NCLS   80
#define HH     160
#define WW     160
#define HWN    (HH*WW)          // 25600 locations per image
#define MPER   (HWN*NCLS)       // 2,048,000 candidates per image
#define KTOP   1000             // PRE_NMS_TOP_N
#define KPAD   1024
#define POSTN  100
#define NBUCKET 4096            // float-bit buckets for top-k threshold
#define CAP    4096             // compacted candidates per image (sorted in LDS)
#define NMS_TH 0.6f
#define PRE_TH 0.05f
#define MINSZ  0.0f

typedef __attribute__((ext_vector_type(2)))  float    v2f;
typedef __attribute__((ext_vector_type(8)))  float    v8f;
typedef __attribute__((ext_vector_type(16))) _Float16 v16h;

// ---------------- workspace layout (bytes) ----------------
#define OFF_HIST   0u           // u32 [8][4096]      = 131072
#define OFF_THR    131072u      // u32 [8]
#define OFF_CNT    131104u      // u32 [8]
#define OFF_CVAL   131136u      // f32 [8][4096]
#define OFF_CIDX   262208u      // u32 [8][4096]
#define OFF_BOX    393280u      // f32 [8][1024][4]
#define OFF_SCORE  524352u      // f32 [8][1024]
#define OFF_LAB    557120u      // u32 [8][1024]
#define OFF_AREA   589888u      // f32 [8][1024]
#define OFF_ADJ    622656u      // u32 [8][1024][32]  = 1 MB adjacency bitmatrix
// total ~1.6 MB

__device__ inline unsigned ballot32(bool p) {
#if __has_builtin(__builtin_amdgcn_ballot_w32)
  return __builtin_amdgcn_ballot_w32(p);
#else
  return (unsigned)__ballot(p);
#endif
}

__device__ inline float sigmoidf(float x) { return 1.0f / (1.0f + expf(-x)); }

// ---------------- K0: zero hist/cnt/adjacency and d_out ----------------
__global__ void k0_init(uint32_t* ws, float* out, int outn) {
  int g = blockIdx.x * blockDim.x + threadIdx.x;
  int stride = gridDim.x * blockDim.x;
  for (int i = g; i < NIMG * NBUCKET; i += stride) ws[OFF_HIST/4 + i] = 0u;
  for (int i = g; i < NIMG;           i += stride) ws[OFF_CNT/4  + i] = 0u;
  for (int i = g; i < NIMG * KPAD * 32; i += stride) ws[OFF_ADJ/4 + i] = 0u;
  for (int i = g; i < outn; i += stride) out[i] = 0.0f;
}

// ---------------- K1: per-image histogram of candidate scores ----------------
// score s = sigmoid(cls)*sigmoid(ctr) in (0,1); float bits are monotonic for
// positive floats, so bucket = bits >> 18 gives a monotone 4096-bucket key.
__global__ __launch_bounds__(256) void k1_hist(const float* __restrict__ cls,
                                               const float* __restrict__ ctr,
                                               uint32_t* ws) {
  __shared__ uint32_t lh[NBUCKET];
  int n = blockIdx.y;
  for (int i = threadIdx.x; i < NBUCKET; i += blockDim.x) lh[i] = 0u;
  __syncthreads();
  const float* clsN = cls + (size_t)n * NCLS * HWN;
  const float* ctrN = ctr + (size_t)n * HWN;
  int g = blockIdx.x * blockDim.x + threadIdx.x;
  int stride = gridDim.x * blockDim.x;
  for (int e = g; e < MPER; e += stride) {       // e = c*HWN + loc (memory order)
    int c = e / HWN, loc = e - c * HWN;
    float sc = sigmoidf(clsN[(size_t)c * HWN + loc]);
    if (sc > PRE_TH) {
      float s = sc * sigmoidf(ctrN[loc]);
      uint32_t b = __float_as_uint(s) >> 18;
      if (b >= NBUCKET) b = NBUCKET - 1;
      atomicAdd(&lh[b], 1u);
    }
  }
  __syncthreads();
  uint32_t* gh = ws + OFF_HIST/4 + n * NBUCKET;
  for (int i = threadIdx.x; i < NBUCKET; i += blockDim.x)
    if (lh[i]) atomicAdd(&gh[i], lh[i]);
}

// ---------------- K2: find threshold bucket for top-1000 ----------------
__global__ void k2_thresh(uint32_t* ws) {
  int n = blockIdx.x;
  if (threadIdx.x != 0) return;
  const uint32_t* gh = ws + OFF_HIST/4 + n * NBUCKET;
  uint32_t cum = 0, T = 0;
  for (int b = NBUCKET - 1; b >= 0; --b) {
    cum += gh[b];
    if (cum >= KTOP) { T = (uint32_t)b; break; }
  }
  ws[OFF_THR/4 + n] = T;
}

// ---------------- K3: compact candidates with bucket >= T ----------------
__global__ __launch_bounds__(256) void k3_compact(const float* __restrict__ cls,
                                                  const float* __restrict__ ctr,
                                                  uint32_t* ws) {
  int n = blockIdx.y;
  uint32_t T = ws[OFF_THR/4 + n];
  float*    cval = (float*)ws + OFF_CVAL/4 + n * CAP;
  uint32_t* cidx = ws + OFF_CIDX/4 + n * CAP;
  uint32_t* cnt  = ws + OFF_CNT/4  + n;
  const float* clsN = cls + (size_t)n * NCLS * HWN;
  const float* ctrN = ctr + (size_t)n * HWN;
  int g = blockIdx.x * blockDim.x + threadIdx.x;
  int stride = gridDim.x * blockDim.x;
  for (int e = g; e < MPER; e += stride) {
    int c = e / HWN, loc = e - c * HWN;
    float sc = sigmoidf(clsN[(size_t)c * HWN + loc]);
    if (sc > PRE_TH) {
      float s = sc * sigmoidf(ctrN[loc]);
      uint32_t b = __float_as_uint(s) >> 18;
      if (b >= NBUCKET) b = NBUCKET - 1;
      if (b >= T) {
        uint32_t p = atomicAdd(cnt, 1u);
        if (p < CAP) {
          cval[p] = s;
          cidx[p] = (uint32_t)loc * NCLS + (uint32_t)c;   // reference flat idx
        }
      }
    }
  }
}

// ---------------- K4: bitonic sort (val desc, idx asc) + box decode ----------
__global__ __launch_bounds__(1024) void k4_sort(const float* __restrict__ reg,
                                                const float* __restrict__ lp,
                                                const float* __restrict__ imsz,
                                                uint32_t* ws) {
  __shared__ unsigned long long key[CAP];       // 32 KB
  int n = blockIdx.x;
  uint32_t cntc = ws[OFF_CNT/4 + n];
  if (cntc > CAP) cntc = CAP;
  const float*    cval = (const float*)ws + OFF_CVAL/4 + n * CAP;
  const uint32_t* cidx = ws + OFF_CIDX/4 + n * CAP;
  for (int i = threadIdx.x; i < CAP; i += blockDim.x) {
    unsigned long long kk = 0ull;                // pads sort to the end
    if (i < (int)cntc)
      kk = ((unsigned long long)__float_as_uint(cval[i]) << 32)
         | (unsigned long long)(0xFFFFFFFFu - cidx[i]);  // tie-break: idx asc
    key[i] = kk;
  }
  __syncthreads();
  for (unsigned k = 2; k <= CAP; k <<= 1)
    for (unsigned j = k >> 1; j > 0; j >>= 1) {
      for (unsigned t = threadIdx.x; t < CAP; t += blockDim.x) {
        unsigned ixj = t ^ j;
        if (ixj > t) {
          unsigned long long a = key[t], b = key[ixj];
          bool up = ((t & k) == 0);
          if (up ? (a < b) : (a > b)) { key[t] = b; key[ixj] = a; }  // descending
        }
      }
      __syncthreads();
    }
  // decode top KPAD slots (slots >= min(cnt,1000) are invalid)
  int r = threadIdx.x;
  int nval = (int)cntc; if (nval > KTOP) nval = KTOP;
  float*    Bx = (float*)ws + OFF_BOX/4   + (size_t)n * KPAD * 4;
  float*    Sc = (float*)ws + OFF_SCORE/4 + n * KPAD;
  uint32_t* Lb = ws + OFF_LAB/4 + n * KPAD;
  float*    Ar = (float*)ws + OFF_AREA/4  + n * KPAD;
  float x1 = 0, y1 = 0, x2 = 0, y2 = 0, score = -1.0f, area = 0.0f;
  uint32_t lab = 0;
  if (r < nval) {
    unsigned long long kk = key[r];
    float    val = __uint_as_float((uint32_t)(kk >> 32));
    uint32_t idx = 0xFFFFFFFFu - (uint32_t)(kk & 0xFFFFFFFFull);
    uint32_t loc = idx / NCLS, c = idx - loc * NCLS;
    lab = c + 1;
    float px = lp[2 * loc], py = lp[2 * loc + 1];
    const float* rg = reg + (size_t)n * 4 * HWN;
    float d0 = rg[0 * HWN + loc], d1 = rg[1 * HWN + loc];
    float d2 = rg[2 * HWN + loc], d3 = rg[3 * HWN + loc];
    float wim = imsz[n * 2 + 0] - 1.0f, him = imsz[n * 2 + 1] - 1.0f;
    x1 = fminf(fmaxf(px - d0, 0.0f), wim);
    y1 = fminf(fmaxf(py - d1, 0.0f), him);
    x2 = fminf(fmaxf(px + d2, 0.0f), wim);
    y2 = fminf(fmaxf(py + d3, 0.0f), him);
    bool valid = (val >= 0.0f) && (x2 - x1 + 1.0f >= MINSZ) && (y2 - y1 + 1.0f >= MINSZ);
    score = valid ? sqrtf(fmaxf(val, 0.0f)) : -1.0f;
    area = (x2 - x1) * (y2 - y1);
  }
  Bx[r * 4 + 0] = x1; Bx[r * 4 + 1] = y1; Bx[r * 4 + 2] = x2; Bx[r * 4 + 3] = y2;
  Sc[r] = score; Lb[r] = lab; Ar[r] = area;
}

// ---------------- K5: WMMA-tiled 1024x1024 IoU adjacency bitmatrix ----------
// union(i,j) = area_i + area_j - inter(i,j).  The pairwise sum area_i+area_j is
// a rank-2 outer product: D = A(16x4) x B(4x16) with A=[area_i, 1, 0, 0] and
// B=[1; area_j; 0; 0] -> one v_wmma_f32_16x16x4_f32 per 16x16 tile.
// Suppression test is division-free:  iou > th  <=>  inter > th * union
// (union clamped >= 1e-6 exactly as the reference clamps the denominator).
// Per-class NMS on offset boxes == (label_i==label_j) && IoU(box)>th, exactly.
__global__ __launch_bounds__(256) void k5_adj(uint32_t* ws) {
  __shared__ float    sbox[KPAD][4];  // 16 KB
  __shared__ float    sarea[KPAD];    //  4 KB
  __shared__ uint32_t slab[KPAD];     //  4 KB
  int n = blockIdx.y;
  const float*    Bx = (const float*)ws + OFF_BOX/4  + (size_t)n * KPAD * 4;
  const float*    Ar = (const float*)ws + OFF_AREA/4 + n * KPAD;
  const uint32_t* Lb = ws + OFF_LAB/4 + n * KPAD;
  for (int i = threadIdx.x; i < KPAD; i += 256) {
    sbox[i][0] = Bx[i * 4 + 0]; sbox[i][1] = Bx[i * 4 + 1];
    sbox[i][2] = Bx[i * 4 + 2]; sbox[i][3] = Bx[i * 4 + 3];
    sarea[i] = Ar[i]; slab[i] = Lb[i];
  }
  __syncthreads();
  uint32_t* adj = ws + OFF_ADJ/4 + (size_t)n * KPAD * 32;
  int lane = threadIdx.x & 31;
  int wave = threadIdx.x >> 5;
  int wg   = blockIdx.x * 8 + wave;           // 128 waves/image, 32 tiles each
  for (int t = wg * 32; t < wg * 32 + 32; ++t) {
    int ti = t >> 6, tj = t & 63;
    float uscale;
    v8f c = {};
#if __has_builtin(__builtin_amdgcn_wmma_f32_16x16x4_f32)
    // A(16x4) f32 layout: lanes 0-15 hold K=0,1; lanes 16-31 hold K=2,3 (zero).
    v2f a, b;
    a.x = (lane < 16) ? sarea[ti * 16 + lane] : 0.0f;
    a.y = (lane < 16) ? 1.0f : 0.0f;
    b.x = (lane < 16) ? 1.0f : 0.0f;
    b.y = (lane < 16) ? sarea[tj * 16 + lane] : 0.0f;
    c = __builtin_amdgcn_wmma_f32_16x16x4_f32(false, a, false, b,
                                              (short)0, c, false, false);
    uscale = 1.0f;
#else
    // fallback: codegen-confirmed f16 WMMA with 2^-16 scaled areas
    const float S = 1.0f / 65536.0f;
    v16h ah = {}, bh = {};
    if (lane < 16) {
      ah[0] = (_Float16)(sarea[ti * 16 + lane] * S); ah[1] = (_Float16)1.0f;
      bh[0] = (_Float16)1.0f; bh[1] = (_Float16)(sarea[tj * 16 + lane] * S);
    }
    c = __builtin_amdgcn_wmma_f32_16x16x32_f16(false, ah, false, bh,
                                               (short)0, c, false, false);
    uscale = 65536.0f;
#endif
    union { v8f v; float f[8]; } du; du.v = c;
#pragma unroll
    for (int r = 0; r < 8; ++r) {
      int m = (lane < 16) ? r : (r + 8);      // C/D layout: lanes16-31 -> M+8
      int i = ti * 16 + m;
      int j = tj * 16 + (lane & 15);
      float ix1 = fmaxf(sbox[i][0], sbox[j][0]);
      float iy1 = fmaxf(sbox[i][1], sbox[j][1]);
      float ix2 = fminf(sbox[i][2], sbox[j][2]);
      float iy2 = fminf(sbox[i][3], sbox[j][3]);
      float inter = fmaxf(ix2 - ix1, 0.0f) * fmaxf(iy2 - iy1, 0.0f);
      float uni   = fmaxf(du.f[r] * uscale - inter, 1e-6f);  // WMMA union sum
      // division-free threshold test: inter/uni > th  <=>  inter > th*uni
      bool pred = (slab[i] == slab[j]) && (slab[i] != 0u) && (inter > NMS_TH * uni);
      unsigned bm = ballot32(pred);            // low16: row ti*16+r, hi16: +8
      if (lane == 0)
        atomicOr(&adj[(size_t)(ti * 16 + r) * 32 + (tj >> 1)],
                 (bm & 0xFFFFu) << ((tj & 1) * 16));
      if (lane == 16)
        atomicOr(&adj[(size_t)(ti * 16 + r + 8) * 32 + (tj >> 1)],
                 (bm >> 16) << ((tj & 1) * 16));
    }
  }
}

// ---------------- K6: greedy bitmask NMS + ranked top-100 output ------------
__global__ __launch_bounds__(1024) void k6_nms(uint32_t* ws, float* out) {
  __shared__ uint32_t kw[32];
  __shared__ uint32_t wp[32];
  int n = blockIdx.x;
  const float*    Sc = (const float*)ws + OFF_SCORE/4 + n * KPAD;
  const float*    Bx = (const float*)ws + OFF_BOX/4   + (size_t)n * KPAD * 4;
  const uint32_t* Lb = ws + OFF_LAB/4 + n * KPAD;
  const uint32_t* adj = ws + OFF_ADJ/4 + (size_t)n * KPAD * 32;
  int tid = threadIdx.x;
  if (tid < 32) {
    uint32_t bits = 0;
    for (int b = 0; b < 32; ++b) {
      int j = tid * 32 + b;
      if (j < KTOP && Sc[j] >= 0.0f) bits |= (1u << b);
    }
    kw[tid] = bits;
  }
  __syncthreads();
  if (tid < 32) {                      // single wave32: lockstep greedy pass
    int lane = tid;
    uint32_t my = kw[lane];
    for (int i = 0; i < KTOP; ++i) {
      int w = i >> 5, bp = i & 31;
      __builtin_prefetch(&adj[(size_t)(i + 1) * 32 + lane], 0, 0);  // hide L2 lat
      uint32_t wi = (uint32_t)__shfl((int)my, w, 32);               // broadcast
      if ((wi >> bp) & 1u) {
        uint32_t arow = adj[(size_t)i * 32 + lane];
        uint32_t mask = (lane > w) ? 0xFFFFFFFFu
                      : (lane == w ? (bp < 31 ? (0xFFFFFFFFu << (bp + 1)) : 0u)
                                   : 0u);                            // only j > i
        my &= ~(arow & mask);
      }
    }
    kw[lane] = my;
  }
  __syncthreads();
  if (tid == 0) {
    uint32_t acc = 0;
    for (int w = 0; w < 32; ++w) { wp[w] = acc; acc += __popc(kw[w]); }
  }
  __syncthreads();
  int j = tid;
  uint32_t w = j >> 5, bp = j & 31;
  if ((kw[w] >> bp) & 1u) {
    uint32_t rank = wp[w] + __popc(kw[w] & (bp ? ((1u << bp) - 1u) : 0u));
    if (rank < POSTN) {                // scores already sorted desc
      float* det = out + (size_t)n * POSTN * 5 + rank * 5;
      det[0] = Bx[j * 4 + 0]; det[1] = Bx[j * 4 + 1];
      det[2] = Bx[j * 4 + 2]; det[3] = Bx[j * 4 + 3];
      det[4] = Sc[j];
      out[NIMG * POSTN * 5 + n * POSTN + rank] = (float)Lb[j];            // label
      out[NIMG * POSTN * 5 + NIMG * POSTN + n * POSTN + rank] = 1.0f;     // valid
    }
  }
}

// ---------------- launch ----------------
extern "C" void kernel_launch(void* const* d_in, const int* in_sizes, int n_in,
                              void* d_out, int out_size, void* d_ws, size_t ws_size,
                              hipStream_t stream) {
  const float* cls  = (const float*)d_in[0];   // (8,80,160,160)
  const float* reg  = (const float*)d_in[1];   // (8,4,160,160)
  const float* ctr  = (const float*)d_in[2];   // (8,1,160,160)
  const float* lp   = (const float*)d_in[3];   // (25600,2)
  const float* imsz = (const float*)d_in[4];   // (8,2)
  uint32_t* ws = (uint32_t*)d_ws;
  float* out = (float*)d_out;

  k0_init   <<<dim3(512),    dim3(256),  0, stream>>>(ws, out, out_size);
  k1_hist   <<<dim3(64, 8),  dim3(256),  0, stream>>>(cls, ctr, ws);
  k2_thresh <<<dim3(8),      dim3(32),   0, stream>>>(ws);
  k3_compact<<<dim3(64, 8),  dim3(256),  0, stream>>>(cls, ctr, ws);
  k4_sort   <<<dim3(8),      dim3(1024), 0, stream>>>(reg, lp, imsz, ws);
  k5_adj    <<<dim3(16, 8),  dim3(256),  0, stream>>>(ws);
  k6_nms    <<<dim3(8),      dim3(1024), 0, stream>>>(ws, out);
}
